// KGSLomics_15135464751344
// MI455X (gfx1250) — matrix operand order repaired and checked by reading, not compiled, not run on hardware
//
#include <hip/hip_runtime.h>
#include <hip/hip_bf16.h>

// ---------------------------------------------------------------------------
// CDNA5 (gfx1250) RGAT implementation.
// GEMMs (node transform einsum, skip MLP) use V_WMMA_F32_16X16X32_BF16
// (bf16 A/B, fp32 accumulate). Edge attention / segment softmax / scatter
// use VALU + global float atomics (the memory-bound part of this workload).
// ---------------------------------------------------------------------------

typedef __attribute__((ext_vector_type(16))) __bf16 v16bf;
typedef __attribute__((ext_vector_type(8)))  float  v8f;

#define LRELU01(v) ((v) > 0.0f ? (v) : 0.01f * (v))
#define LRELU20(v) ((v) > 0.0f ? (v) : 0.2f  * (v))

__device__ __forceinline__ unsigned short f2bf(float f) {
  unsigned u = __float_as_uint(f);
  unsigned r = u + 0x7FFFu + ((u >> 16) & 1u);   // round-to-nearest-even
  return (unsigned short)(r >> 16);
}

__device__ __forceinline__ void atomicMaxF(float* addr, float val) {
  unsigned* ua = (unsigned*)addr;
  unsigned cur = __float_as_uint(*addr);
  while (__uint_as_float(cur) < val) {
    unsigned prev = atomicCAS(ua, cur, __float_as_uint(val));
    if (prev == cur) break;
    cur = prev;
  }
}

union BFrag { uint4 u[2]; v16bf v; };

// ---------------------------------------------------------------------------
// Repack W[r][256][256] (row-major d,o) into WMMA B-fragment order:
//   Wp[((r*8+kt)*16+nt)*32 + lane][t] = W[r][kt*32 + (lane>>4)*8 + (t&7) + (t>>3)*16][nt*16 + (lane&15)]
// so each lane's B fragment is 16 contiguous bf16 (two b128 loads).
// ---------------------------------------------------------------------------
__global__ void pack_wmma_b(const float* __restrict__ W,
                            unsigned short* __restrict__ Wp, int nrel) {
  long idx = (long)blockIdx.x * blockDim.x + threadIdx.x;
  long total = (long)nrel << 16;
  if (idx >= total) return;
  int r    = (int)(idx >> 16);
  int rem  = (int)(idx & 65535);
  int t    = rem & 15;
  int lane = (rem >> 4) & 31;
  int nt   = (rem >> 9) & 15;
  int kt   = (rem >> 13) & 7;
  int n  = nt * 16 + (lane & 15);
  int kk = kt * 32 + (lane >> 4) * 8 + (t & 7) + (t >> 3) * 16;
  Wp[idx] = f2bf(W[((long)r << 16) + (long)kk * 256 + n]);
}

// ---------------------------------------------------------------------------
// ccle MLP + gather + concat -> bf16 x_in [N,256]
// ---------------------------------------------------------------------------
__global__ void build_xin(const float* __restrict__ kg,
                          const float* __restrict__ ccle,
                          const int* __restrict__ node_id,
                          const float* __restrict__ cw1, const float* __restrict__ cb1,
                          const float* __restrict__ cw2, const float* __restrict__ cb2,
                          unsigned short* __restrict__ xinb, int n) {
  __shared__ float s_w1[4 * 32];
  __shared__ float s_b1[32];
  __shared__ float s_w2[32 * 128];
  __shared__ float s_b2[128];
  for (int i = threadIdx.x; i < 128;  i += blockDim.x) s_w1[i] = cw1[i];
  for (int i = threadIdx.x; i < 32;   i += blockDim.x) s_b1[i] = cb1[i];
  for (int i = threadIdx.x; i < 4096; i += blockDim.x) s_w2[i] = cw2[i];
  for (int i = threadIdx.x; i < 128;  i += blockDim.x) s_b2[i] = cb2[i];
  __syncthreads();
  int i = blockIdx.x * blockDim.x + threadIdx.x;
  if (i >= n) return;
  int nid = node_id[i];
  const float* cr = ccle + (long)nid * 4;
  float c0 = cr[0], c1 = cr[1], c2 = cr[2], c3 = cr[3];
  float h[32];
#pragma unroll
  for (int o = 0; o < 32; ++o) {
    float v = s_b1[o] + c0 * s_w1[o] + c1 * s_w1[32 + o] +
              c2 * s_w1[64 + o] + c3 * s_w1[96 + o];
    h[o] = LRELU01(v);
  }
  const float* kr = kg + (long)nid * 128;
  unsigned short* xr = xinb + (long)i * 256;
  for (int j = 0; j < 128; ++j) xr[j] = f2bf(kr[j]);
#pragma unroll 4
  for (int o = 0; o < 128; ++o) {
    float v = s_b2[o];
#pragma unroll
    for (int t = 0; t < 32; ++t) v += h[t] * s_w2[t * 128 + o];
    xr[128 + o] = f2bf(v);
  }
}

// ---------------------------------------------------------------------------
// WMMA GEMM: Y[r][i][o] = sum_d Xb[i][d] * W[r][d][o]   (D = O = 256)
// One wave computes a 16x64 strip (4 C tiles); block = 8 waves = 128x64.
// Grid: ((nrows+127)/128, 4, nrel).
// Optional fused bias / leaky-relu(0.01) / bf16 output.
// ---------------------------------------------------------------------------
__global__ void wmma_gemm256(const unsigned short* __restrict__ Xb,
                             const unsigned short* __restrict__ Wp,
                             float* __restrict__ Yf,
                             unsigned short* __restrict__ Ybf,
                             const float* __restrict__ bias,
                             int nrows, int apply_lrelu, long relStrideY) {
  const int lane    = threadIdx.x & 31;
  const int wv      = threadIdx.x >> 5;
  const int r       = blockIdx.z;
  const int rowBase = blockIdx.x * 128 + wv * 16;
  const int colBase = blockIdx.y * 64;
  const unsigned short* Wr = Wp + ((long)r << 16);

  // A fragment addressing (ISA 16-bit A 16x32 layout): row = lane&15,
  // per-lane K groups start at (lane>>4)*8 and 16+(lane>>4)*8.
  int m  = lane & 15;
  int kb = (lane >> 4) * 8;
  int arow = rowBase + m;
  if (arow >= nrows) arow = nrows - 1;     // clamp: keep EXEC all-1s for WMMA
  const unsigned short* aptr = Xb + (long)arow * 256 + kb;

  v8f zero = {0.f, 0.f, 0.f, 0.f, 0.f, 0.f, 0.f, 0.f};
  v8f acc0 = zero, acc1 = zero, acc2 = zero, acc3 = zero;

  const int ntBase = colBase >> 4;
#pragma unroll
  for (int kt = 0; kt < 8; ++kt) {
    BFrag a;
    a.u[0] = *(const uint4*)(aptr + kt * 32);
    a.u[1] = *(const uint4*)(aptr + kt * 32 + 16);
    if (kt < 7) __builtin_prefetch(aptr + (kt + 1) * 32, 0, 1);

    const unsigned short* brow =
        Wr + ((((long)kt * 16 + ntBase) * 32 + lane) << 4);
    BFrag b0, b1, b2, b3;
    b0.u[0] = *(const uint4*)(brow);          b0.u[1] = *(const uint4*)(brow + 8);
    b1.u[0] = *(const uint4*)(brow + 512);    b1.u[1] = *(const uint4*)(brow + 520);
    b2.u[0] = *(const uint4*)(brow + 1024);   b2.u[1] = *(const uint4*)(brow + 1032);
    b3.u[0] = *(const uint4*)(brow + 1536);   b3.u[1] = *(const uint4*)(brow + 1544);

    acc0 = __builtin_amdgcn_wmma_f32_16x16x32_bf16(false, a.v, false, b0.v,
                                                   (short)0, acc0, false, false);
    acc1 = __builtin_amdgcn_wmma_f32_16x16x32_bf16(false, a.v, false, b1.v,
                                                   (short)0, acc1, false, false);
    acc2 = __builtin_amdgcn_wmma_f32_16x16x32_bf16(false, a.v, false, b2.v,
                                                   (short)0, acc2, false, false);
    acc3 = __builtin_amdgcn_wmma_f32_16x16x32_bf16(false, a.v, false, b3.v,
                                                   (short)0, acc3, false, false);
  }

  // C/D layout: VGPR g -> row rowBase + (lane>>4)*8 + g, col tile + (lane&15)
  int rbase = rowBase + (lane >> 4) * 8;
  int cbase = colBase + (lane & 15);
#pragma unroll
  for (int g = 0; g < 8; ++g) {
    int rr = rbase + g;
    if (rr < nrows) {
      float vv[4] = {acc0[g], acc1[g], acc2[g], acc3[g]};
#pragma unroll
      for (int c = 0; c < 4; ++c) {
        float v = vv[c];
        int cc = cbase + c * 16;
        if (bias) v += bias[cc];
        if (apply_lrelu) v = LRELU01(v);
        long off = relStrideY * (long)r + (long)rr * 256 + cc;
        if (Yf)  Yf[off]  = v;
        if (Ybf) Ybf[off] = f2bf(v);
      }
    }
  }
}

// ---------------------------------------------------------------------------
// Per-edge attention logits: one wave per edge, 4 heads, leaky(0.2).
// Also segment-max into mbuf via CAS float-max.
// ---------------------------------------------------------------------------
__global__ void attn_logits(const float* __restrict__ xw,
                            const int* __restrict__ src, const int* __restrict__ dst,
                            const int* __restrict__ ety,
                            const float* __restrict__ q, const float* __restrict__ k,
                            float* __restrict__ logits, float* __restrict__ mbuf,
                            int nE, int nrows) {
  int lane = threadIdx.x & 31;
  int e = blockIdx.x * 8 + (threadIdx.x >> 5);
  if (e >= nE) return;
  int sn = src[e], dn = dst[e], t = ety[e];
  const float* oi = xw + ((long)t * nrows + dn) * 256;  // target feats
  const float* oj = xw + ((long)t * nrows + sn) * 256;  // source feats
  float a0 = 0.f, a1 = 0.f, a2 = 0.f, a3 = 0.f;
#pragma unroll
  for (int c0 = 0; c0 < 8; ++c0) {
    int c = c0 * 32 + lane;
    float xi = oi[c], xj = oj[c];
    const float* qr = q + (long)c * 4;
    const float* kr = k + (long)c * 4;
    a0 += xi * qr[0] + xj * kr[0];
    a1 += xi * qr[1] + xj * kr[1];
    a2 += xi * qr[2] + xj * kr[2];
    a3 += xi * qr[3] + xj * kr[3];
  }
#pragma unroll
  for (int off = 16; off > 0; off >>= 1) {   // wave32 reduction
    a0 += __shfl_xor(a0, off, 32);
    a1 += __shfl_xor(a1, off, 32);
    a2 += __shfl_xor(a2, off, 32);
    a3 += __shfl_xor(a3, off, 32);
  }
  if (lane < 4) {
    float v = lane == 0 ? a0 : lane == 1 ? a1 : lane == 2 ? a2 : a3;
    v = LRELU20(v);
    logits[(long)e * 4 + lane] = v;
    atomicMaxF(&mbuf[(long)dn * 4 + lane], v);
  }
}

__global__ void attn_expsum(float* __restrict__ logits,
                            const int* __restrict__ dst,
                            const float* __restrict__ mbuf,
                            float* __restrict__ sbuf, long nE4) {
  long i = (long)blockIdx.x * blockDim.x + threadIdx.x;
  if (i >= nE4) return;
  long e = i >> 2;
  int  h = (int)(i & 3);
  int  d = dst[e];
  float mm = mbuf[(long)d * 4 + h];
  float ev = __expf(logits[i] - mm);
  logits[i] = ev;
  atomicAdd(&sbuf[(long)d * 4 + h], ev);
}

__global__ void attn_aggregate(const float* __restrict__ xw,
                               const float* __restrict__ eA,
                               const float* __restrict__ sbuf,
                               const int* __restrict__ src, const int* __restrict__ dst,
                               const int* __restrict__ ety,
                               float* __restrict__ agg, int nrows) {
  int e = blockIdx.x;
  int c = threadIdx.x;                         // 0..255
  int sn = src[e], dn = dst[e], t = ety[e];
  int h = c >> 6;                              // head = channel/64
  float den  = sbuf[(long)dn * 4 + h];
  float coef = eA[(long)e * 4 + h] / fmaxf(den, 1e-16f);
  float v = xw[((long)t * nrows + sn) * 256 + c] * coef;
  atomicAdd(&agg[(long)dn * 256 + c], v);
}

// ---------------------------------------------------------------------------
// small utility kernels
// ---------------------------------------------------------------------------
__global__ void fill_f32(float* __restrict__ p, float v, long n) {
  long i = (long)blockIdx.x * blockDim.x + threadIdx.x;
  if (i < n) p[i] = v;
}

__global__ void bias_lrelu_tobf(const float* __restrict__ agg,
                                const float* __restrict__ bias,
                                unsigned short* __restrict__ ob, long n256) {
  long i = (long)blockIdx.x * blockDim.x + threadIdx.x;
  if (i >= n256) return;
  int c = (int)(i & 255);
  float v = agg[i] + bias[c];
  v = LRELU01(v);
  ob[i] = f2bf(v);
}

__global__ void final_out(const float* __restrict__ agg,
                          const float* __restrict__ bias,
                          const float* __restrict__ skip2,
                          float* __restrict__ out, long n256) {
  long i = (long)blockIdx.x * blockDim.x + threadIdx.x;
  if (i >= n256) return;
  int c = (int)(i & 255);
  float v = agg[i] + bias[c] + skip2[i];
  out[i] = LRELU01(v);
}

// ---------------------------------------------------------------------------
extern "C" void kernel_launch(void* const* d_in, const int* in_sizes, int n_in,
                              void* d_out, int out_size, void* d_ws, size_t ws_size,
                              hipStream_t stream) {
  const float* kg      = (const float*)d_in[0];
  const float* ccle    = (const float*)d_in[1];
  const int*   node_id = (const int*)d_in[2];
  const int*   eidx    = (const int*)d_in[3];
  const int*   ety     = (const int*)d_in[4];
  const float* cw1 = (const float*)d_in[5];
  const float* cb1 = (const float*)d_in[6];
  const float* cw2 = (const float*)d_in[7];
  const float* cb2 = (const float*)d_in[8];
  const float* w1  = (const float*)d_in[9];
  const float* q1  = (const float*)d_in[10];
  const float* k1  = (const float*)d_in[11];
  const float* b1  = (const float*)d_in[12];
  const float* w2  = (const float*)d_in[13];
  const float* q2  = (const float*)d_in[14];
  const float* k2  = (const float*)d_in[15];
  const float* b2  = (const float*)d_in[16];
  const float* sw1 = (const float*)d_in[17];
  const float* sb1 = (const float*)d_in[18];
  const float* sw2 = (const float*)d_in[19];
  const float* sb2 = (const float*)d_in[20];
  float* out = (float*)d_out;
  (void)n_in; (void)out_size; (void)ws_size;

  const int N = in_sizes[2];               // nodes
  const int E = in_sizes[4];               // edges
  const int R = in_sizes[9] / (256 * 256); // relations
  const int* src = eidx;
  const int* dst = eidx + E;

  // workspace carve (all buffers fully written each launch)
  char* p = (char*)d_ws;
  auto take = [&](size_t bytes) {
    char* q = p;
    p += (bytes + 255) & ~(size_t)255;
    return (void*)q;
  };
  unsigned short* xinb = (unsigned short*)take((size_t)N * 256 * 2);
  unsigned short* x1b  = (unsigned short*)take((size_t)N * 256 * 2);
  unsigned short* s1b  = (unsigned short*)take((size_t)N * 256 * 2);
  unsigned short* w1p  = (unsigned short*)take((size_t)R * 65536 * 2);
  unsigned short* w2p  = (unsigned short*)take((size_t)R * 65536 * 2);
  unsigned short* sw1p = (unsigned short*)take((size_t)65536 * 2);
  unsigned short* sw2p = (unsigned short*)take((size_t)65536 * 2);
  float* xw     = (float*)take((size_t)R * N * 256 * 4);
  float* logits = (float*)take((size_t)E * 4 * 4);
  float* mbuf   = (float*)take((size_t)N * 4 * 4);
  float* sbuf   = (float*)take((size_t)N * 4 * 4);
  float* agg    = (float*)take((size_t)N * 256 * 4);
  float* skip2  = (float*)take((size_t)N * 256 * 4);

  // ---- weight repack to WMMA B-fragment layout (bf16) ----
  pack_wmma_b<<<(R * 65536 + 255) / 256, 256, 0, stream>>>(w1, w1p, R);
  pack_wmma_b<<<(R * 65536 + 255) / 256, 256, 0, stream>>>(w2, w2p, R);
  pack_wmma_b<<<(65536 + 255) / 256, 256, 0, stream>>>(sw1, sw1p, 1);
  pack_wmma_b<<<(65536 + 255) / 256, 256, 0, stream>>>(sw2, sw2p, 1);

  // ---- input features ----
  build_xin<<<(N + 127) / 128, 128, 0, stream>>>(kg, ccle, node_id,
                                                 cw1, cb1, cw2, cb2, xinb, N);

  dim3 ggrid((N + 127) / 128, 4, R);
  dim3 sgrid((N + 127) / 128, 4, 1);

  // xw[r] = x_in @ w1[r]   (WMMA bf16 -> f32)
  wmma_gemm256<<<ggrid, 256, 0, stream>>>(xinb, w1p, xw, nullptr, nullptr,
                                          N, 0, (long)N * 256);
  // skip path: skip1(bf16) = lrelu(x_in @ sw1 + sb1); skip2 = skip1 @ sw2 + sb2
  wmma_gemm256<<<sgrid, 256, 0, stream>>>(xinb, sw1p, nullptr, s1b, sb1, N, 1, 0);
  wmma_gemm256<<<sgrid, 256, 0, stream>>>(s1b, sw2p, skip2, nullptr, sb2, N, 0, 0);

  // ---- layer 1 attention ----
  hipMemsetAsync(agg,  0, (size_t)N * 256 * 4, stream);
  hipMemsetAsync(sbuf, 0, (size_t)N * 4 * 4, stream);
  fill_f32<<<(int)(((long)N * 4 + 255) / 256), 256, 0, stream>>>(mbuf, -1e30f,
                                                                 (long)N * 4);
  attn_logits<<<(E + 7) / 8, 256, 0, stream>>>(xw, src, dst, ety, q1, k1,
                                               logits, mbuf, E, N);
  attn_expsum<<<(int)(((long)E * 4 + 255) / 256), 256, 0, stream>>>(
      logits, dst, mbuf, sbuf, (long)E * 4);
  attn_aggregate<<<E, 256, 0, stream>>>(xw, logits, sbuf, src, dst, ety, agg, N);
  bias_lrelu_tobf<<<(int)(((long)N * 256 + 255) / 256), 256, 0, stream>>>(
      agg, b1, x1b, (long)N * 256);

  // ---- layer 2 ----
  hipMemsetAsync(agg,  0, (size_t)N * 256 * 4, stream);
  hipMemsetAsync(sbuf, 0, (size_t)N * 4 * 4, stream);
  fill_f32<<<(int)(((long)N * 4 + 255) / 256), 256, 0, stream>>>(mbuf, -1e30f,
                                                                 (long)N * 4);
  wmma_gemm256<<<ggrid, 256, 0, stream>>>(x1b, w2p, xw, nullptr, nullptr,
                                          N, 0, (long)N * 256);
  attn_logits<<<(E + 7) / 8, 256, 0, stream>>>(xw, src, dst, ety, q2, k2,
                                               logits, mbuf, E, N);
  attn_expsum<<<(int)(((long)E * 4 + 255) / 256), 256, 0, stream>>>(
      logits, dst, mbuf, sbuf, (long)E * 4);
  attn_aggregate<<<E, 256, 0, stream>>>(xw, logits, sbuf, src, dst, ety, agg, N);

  // out = lrelu(agg + bias2 + skip2)
  final_out<<<(int)(((long)N * 256 + 255) / 256), 256, 0, stream>>>(
      agg, b2, skip2, out, (long)N * 256);
}
